// AttentionTalkingHead_32203664785574
// MI455X (gfx1250) — matrix operand order repaired and checked by reading, not compiled, not run on hardware
//
#include <hip/hip_runtime.h>
#include <hip/hip_bf16.h>

#define TB   4
#define TN   2048
#define TDIM 384
#define THD  6
#define TD   64

typedef __attribute__((ext_vector_type(2)))  float  v2f;
typedef __attribute__((ext_vector_type(8)))  float  v8f;
typedef __attribute__((ext_vector_type(8)))  __bf16 v8bf;
typedef __attribute__((ext_vector_type(16))) __bf16 v16bf;

// ---- fp32 WMMA: D(16x16) = A(16x4) * B(4x16) + C -------------------------
// A: lane l holds A[l%16][2*(l/16)+{0,1}]
// B: lane l holds B[2*(l/16)+{0,1}][l%16]
// D: element j in lane l is D[j + 8*(l/16)][l%16]
__device__ __forceinline__ v8f wmma_f32_k4(v2f a, v2f b, v8f c) {
  return __builtin_amdgcn_wmma_f32_16x16x4_f32(false, a, false, b, (short)0, c,
                                               false, false);
}

// ---- bf16 WMMA: D(16x16,f32) = A(16x32,bf16) * B(32x16,bf16) + C ---------
// A: lane l element e: K = 16*(l/16)+e (order used by callers, same as B); M=l%16
// B: lane l element e: K = 16*(l/16)+e ; N = l%16
__device__ __forceinline__ v8f wmma_bf16(v16bf a, v16bf b, v8f c) {
  return __builtin_amdgcn_wmma_f32_16x16x32_bf16(false, a, false, b, (short)0,
                                                 c, false, false);
}

// load A-operand halves: p0 -> elements 0..7, p1 -> elements 8..15 (each 16B)
__device__ __forceinline__ v16bf ld_a16(const __bf16* p0, const __bf16* p1) {
  v8bf x = *(const v8bf*)p0;
  v8bf y = *(const v8bf*)p1;
  v16bf r;
#pragma unroll
  for (int i = 0; i < 8; ++i) {
    r[i] = x[i];
    r[i + 8] = y[i];
  }
  return r;
}

// ---------------------------------------------------------------------------
// Kernel 1: qkv = x @ w_qkv + b_qkv (fp32 WMMA), results converted to bf16:
//   q (scaled by d^-1/2) -> [B][H][N][64],  k -> [B][H][N][64],
//   v -> transposed [B][H][64][N] so PV B-operands are contiguous.
// ---------------------------------------------------------------------------
__global__ __launch_bounds__(256) void k_qkv(const float* __restrict__ x,
                                             const float* __restrict__ w,
                                             const float* __restrict__ bias,
                                             __bf16* __restrict__ qb,
                                             __bf16* __restrict__ kb,
                                             __bf16* __restrict__ vt) {
  const int lane = threadIdx.x & 31;
  const int wave = threadIdx.x >> 5;
  const int r  = lane & 15;
  const int hh = lane >> 4;
  const int n0 = blockIdx.x * 64;
  const int m0 = blockIdx.y * 128 + wave * 16;

  v8f acc[4];
#pragma unroll
  for (int cb = 0; cb < 4; ++cb) {
    float bv = bias[n0 + cb * 16 + r];
#pragma unroll
    for (int j = 0; j < 8; ++j) acc[cb][j] = bv;
  }

  for (int k0 = 0; k0 < TDIM; k0 += 16) {
    v2f a[4];
#pragma unroll
    for (int t = 0; t < 4; ++t) {
      const float* ap = x + (size_t)(m0 + r) * TDIM + (k0 + 4 * t + 2 * hh);
      a[t].x = ap[0];
      a[t].y = ap[1];
    }
#pragma unroll
    for (int t = 0; t < 4; ++t) {
#pragma unroll
      for (int cb = 0; cb < 4; ++cb) {
        const float* bp =
            w + (size_t)(k0 + 4 * t + 2 * hh) * (3 * TDIM) + (n0 + cb * 16 + r);
        v2f bo;
        bo.x = bp[0];
        bo.y = bp[3 * TDIM];
        acc[cb] = wmma_f32_k4(a[t], bo, acc[cb]);
      }
    }
  }

  const int bi    = m0 / TN;
  const int nbase = m0 % TN;
  const int which = n0 / TDIM;          // 0=q 1=k 2=v (strip never crosses)
  const int hd    = (n0 % TDIM) / TD;

  if (which < 2) {
    __bf16* dst    = (which == 0) ? qb : kb;
    const float sc = (which == 0) ? 0.125f : 1.0f;  // 64^-0.5
    const size_t base = (((size_t)bi * THD + hd) * TN) * TD;
#pragma unroll
    for (int cb = 0; cb < 4; ++cb) {
      const int dd = cb * 16 + r;
#pragma unroll
      for (int j = 0; j < 8; ++j) {
        const int n = nbase + j + 8 * hh;
        dst[base + (size_t)n * TD + dd] = (__bf16)(acc[cb][j] * sc);
      }
    }
  } else {
    const size_t base = (((size_t)bi * THD + hd) * TD) * TN;
#pragma unroll
    for (int cb = 0; cb < 4; ++cb) {
      const int dd = cb * 16 + r;
#pragma unroll
      for (int j = 0; j < 8; ++j) {
        const int n = nbase + j + 8 * hh;
        vt[base + (size_t)dd * TN + n] = (__bf16)acc[cb][j];
      }
    }
  }
}

// ---------------------------------------------------------------------------
// Kernel 2: fused talking-heads attention, two-pass flash style, bf16 WMMA.
// Block = 6 waves; wave g owns head g; block owns a 16-row query tile.
// Processes 32 keys per step (K=32 per bf16 WMMA).
// ---------------------------------------------------------------------------
__global__ __launch_bounds__(192) void k_attn(const __bf16* __restrict__ qbuf,
                                              const __bf16* __restrict__ kbuf,
                                              const __bf16* __restrict__ vtbuf,
                                              const float* __restrict__ wl,
                                              const float* __restrict__ blv,
                                              const float* __restrict__ wwv,
                                              const float* __restrict__ bwv,
                                              float* __restrict__ oh) {
  __shared__ float rawS[THD][16][36];  // raw scores, 32 key cols + pad
  __shared__ float pS[THD][16][36];    // normalized probabilities
  __shared__ float stM[THD][16];
  __shared__ float stZ[THD][16];
  __shared__ float sWL[36], sWW[36], sBL[THD], sBW[THD];

  const int lane = threadIdx.x & 31;
  const int g    = threadIdx.x >> 5;  // head / wave id 0..5
  const int r    = lane & 15;
  const int hh   = lane >> 4;
  const int bi   = blockIdx.y;
  const int i0   = blockIdx.x * 16;   // query row tile

  if (threadIdx.x < 36) {
    sWL[threadIdx.x] = wl[threadIdx.x];
    sWW[threadIdx.x] = wwv[threadIdx.x];
  }
  if (threadIdx.x < THD) {
    sBL[threadIdx.x] = blv[threadIdx.x];
    sBW[threadIdx.x] = bwv[threadIdx.x];
  }
  if (lane < 16) {
    stM[g][lane] = -1e30f;
    stZ[g][lane] = 0.0f;
  }
  __syncthreads();

  const size_t hoff = ((size_t)bi * THD + g) * (size_t)TN * TD;
  const __bf16* Q  = qbuf + hoff;
  const __bf16* K  = kbuf + hoff;
  const __bf16* Vt = vtbuf + hoff;  // [64][N] for this (b,g)

  // Q tile (16x64) as two K=32 A-operands, resident all kernel
  const __bf16* Qr = Q + (size_t)(i0 + r) * TD;
  v16bf qa[2];
  qa[0] = ld_a16(Qr + 16 * hh, Qr + 8 + 16 * hh);
  qa[1] = ld_a16(Qr + 32 + 16 * hh, Qr + 40 + 16 * hh);

  // ------------------ pass 1: softmax statistics ------------------
  for (int mb = 0; mb < TN / 32; ++mb) {
    const int m0 = mb * 32;
    const __bf16* K0 = K + (size_t)(m0 + r) * TD;       // keys m0..m0+15
    const __bf16* K1 = K + (size_t)(m0 + 16 + r) * TD;  // keys m0+16..m0+31
    v8f s0, s1;
#pragma unroll
    for (int j = 0; j < 8; ++j) { s0[j] = 0.0f; s1[j] = 0.0f; }
#pragma unroll
    for (int kc = 0; kc < 2; ++kc) {
      v16bf b0 = *(const v16bf*)(K0 + kc * 32 + 16 * hh);
      v16bf b1 = *(const v16bf*)(K1 + kc * 32 + 16 * hh);
      s0 = wmma_bf16(qa[kc], b0, s0);
      s1 = wmma_bf16(qa[kc], b1, s1);
    }
    __syncthreads();  // previous iteration's rawS readers done
#pragma unroll
    for (int j = 0; j < 8; ++j) {
      rawS[g][j + 8 * hh][r] = s0[j];
      rawS[g][j + 8 * hh][16 + r] = s1[j];
    }
    __syncthreads();

    // pre-softmax head mix; lane handles row r, cols [16*hh, 16*hh+16)
    float vals[16];
    float tmax = -1e30f;
#pragma unroll
    for (int cc = 0; cc < 16; ++cc) {
      const int c = 16 * hh + cc;
      float a = sBL[g];
#pragma unroll
      for (int hs = 0; hs < THD; ++hs) a += rawS[hs][r][c] * sWL[hs * THD + g];
      vals[cc] = a;
      tmax = fmaxf(tmax, a);
    }
    tmax = fmaxf(tmax, __shfl_xor(tmax, 16, 32));
    const float runM = stM[g][r];
    const float runZ = stZ[g][r];
    const float newM = fmaxf(runM, tmax);
    float esum = 0.0f;
#pragma unroll
    for (int cc = 0; cc < 16; ++cc) esum += __expf(vals[cc] - newM);
    esum += __shfl_xor(esum, 16, 32);
    if (lane < 16) {
      stZ[g][r] = runZ * __expf(runM - newM) + esum;
      stM[g][r] = newM;
    }
  }
  __syncthreads();

  // ------------------ pass 2: recompute + PV ------------------
  v8f o[4];
#pragma unroll
  for (int cb = 0; cb < 4; ++cb)
#pragma unroll
    for (int j = 0; j < 8; ++j) o[cb][j] = 0.0f;

  for (int mb = 0; mb < TN / 32; ++mb) {
    const int m0 = mb * 32;
    const __bf16* K0 = K + (size_t)(m0 + r) * TD;
    const __bf16* K1 = K + (size_t)(m0 + 16 + r) * TD;
    v8f s0, s1;
#pragma unroll
    for (int j = 0; j < 8; ++j) { s0[j] = 0.0f; s1[j] = 0.0f; }
#pragma unroll
    for (int kc = 0; kc < 2; ++kc) {
      v16bf b0 = *(const v16bf*)(K0 + kc * 32 + 16 * hh);
      v16bf b1 = *(const v16bf*)(K1 + kc * 32 + 16 * hh);
      s0 = wmma_bf16(qa[kc], b0, s0);
      s1 = wmma_bf16(qa[kc], b1, s1);
    }
    __syncthreads();
#pragma unroll
    for (int j = 0; j < 8; ++j) {
      rawS[g][j + 8 * hh][r] = s0[j];
      rawS[g][j + 8 * hh][16 + r] = s1[j];
    }
    __syncthreads();

    const float M  = stM[g][r];
    const float Zi = 1.0f / stZ[g][r];
#pragma unroll
    for (int cc = 0; cc < 16; ++cc) {
      const int c = 16 * hh + cc;
      float a = sBL[g];
#pragma unroll
      for (int hs = 0; hs < THD; ++hs) a += rawS[hs][r][c] * sWL[hs * THD + g];
      pS[g][r][c] = __expf(a - M) * Zi;
    }
    __syncthreads();

    // post-softmax mix directly into the bf16 A-operand layout
    v16bf pa;
#pragma unroll
    for (int e = 0; e < 8; ++e) {
      const int cA = 16 * hh + e;       // keys 16*hh..16*hh+7 of block
      const int cB = 16 * hh + 8 + e;   // keys 16*hh+8..16*hh+15 of block
      float aA = sBW[g], aB = sBW[g];
#pragma unroll
      for (int hs = 0; hs < THD; ++hs) {
        const float wgt = sWW[hs * THD + g];
        aA += pS[hs][r][cA] * wgt;
        aB += pS[hs][r][cB] * wgt;
      }
      pa[e] = (__bf16)aA;
      pa[e + 8] = (__bf16)aB;
    }
    // O += P2 @ V  (V transposed: contiguous per-lane B loads)
#pragma unroll
    for (int cb = 0; cb < 4; ++cb) {
      const __bf16* vp = Vt + (size_t)(cb * 16 + r) * TN + m0 + 16 * hh;
      v16bf vo = *(const v16bf*)vp;
      o[cb] = wmma_bf16(pa, vo, o[cb]);
    }
  }

  // write O into the head-concatenated [B][N][DIM] fp32 layout
  float* dst = oh + (size_t)bi * TN * TDIM;
#pragma unroll
  for (int cb = 0; cb < 4; ++cb)
#pragma unroll
    for (int j = 0; j < 8; ++j)
      dst[(size_t)(i0 + j + 8 * hh) * TDIM + g * TD + cb * 16 + r] = o[cb][j];
}

// ---------------------------------------------------------------------------
// Kernel 3: out = oh @ w_proj + b_proj (fp32 WMMA)
// ---------------------------------------------------------------------------
__global__ __launch_bounds__(256) void k_proj(const float* __restrict__ a_,
                                              const float* __restrict__ w,
                                              const float* __restrict__ bias,
                                              float* __restrict__ out) {
  const int lane = threadIdx.x & 31;
  const int wave = threadIdx.x >> 5;
  const int r  = lane & 15;
  const int hh = lane >> 4;
  const int n0 = blockIdx.x * 64;
  const int m0 = blockIdx.y * 128 + wave * 16;

  v8f acc[4];
#pragma unroll
  for (int cb = 0; cb < 4; ++cb) {
    float bv = bias[n0 + cb * 16 + r];
#pragma unroll
    for (int j = 0; j < 8; ++j) acc[cb][j] = bv;
  }

  for (int k0 = 0; k0 < TDIM; k0 += 16) {
    v2f a[4];
#pragma unroll
    for (int t = 0; t < 4; ++t) {
      const float* ap = a_ + (size_t)(m0 + r) * TDIM + (k0 + 4 * t + 2 * hh);
      a[t].x = ap[0];
      a[t].y = ap[1];
    }
#pragma unroll
    for (int t = 0; t < 4; ++t) {
#pragma unroll
      for (int cb = 0; cb < 4; ++cb) {
        const float* bp =
            w + (size_t)(k0 + 4 * t + 2 * hh) * TDIM + (n0 + cb * 16 + r);
        v2f bo;
        bo.x = bp[0];
        bo.y = bp[TDIM];
        acc[cb] = wmma_f32_k4(a[t], bo, acc[cb]);
      }
    }
  }

#pragma unroll
  for (int cb = 0; cb < 4; ++cb)
#pragma unroll
    for (int j = 0; j < 8; ++j)
      out[(size_t)(m0 + j + 8 * hh) * TDIM + n0 + cb * 16 + r] = acc[cb][j];
}

extern "C" void kernel_launch(void* const* d_in, const int* in_sizes, int n_in,
                              void* d_out, int out_size, void* d_ws,
                              size_t ws_size, hipStream_t stream) {
  const float* x     = (const float*)d_in[0];
  const float* w_qkv = (const float*)d_in[1];
  const float* b_qkv = (const float*)d_in[2];
  const float* w_l   = (const float*)d_in[3];
  const float* b_l   = (const float*)d_in[4];
  const float* w_w   = (const float*)d_in[5];
  const float* b_w   = (const float*)d_in[6];
  const float* w_pr  = (const float*)d_in[7];
  const float* b_pr  = (const float*)d_in[8];
  float* out = (float*)d_out;

  const size_t per = (size_t)TB * THD * TN * TD;  // elements per q/k/v buffer
  __bf16* qb = (__bf16*)d_ws;
  __bf16* kb = qb + per;
  __bf16* vt = kb + per;
  float*  oh = (float*)(vt + per);  // 3*per*2 bytes in, 256B-aligned

  k_qkv<<<dim3(3 * TDIM / 64, (TB * TN) / 128), 256, 0, stream>>>(
      x, w_qkv, b_qkv, qb, kb, vt);
  k_attn<<<dim3(TN / 16, TB), 192, 0, stream>>>(qb, kb, vt, w_l, b_l, w_w, b_w,
                                                oh);
  k_proj<<<dim3(TDIM / 64, (TB * TN) / 128), 256, 0, stream>>>(oh, w_pr, b_pr,
                                                               out);
}
